// SimilarityMatchingModel_10831907520535
// MI455X (gfx1250) — matrix operand order repaired and checked by reading, not compiled, not run on hardware
//
#include <hip/hip_runtime.h>
#include <hip/hip_bf16.h>
#include <stdint.h>

// ---------------------------------------------------------------------------
// MI455X (gfx1250) implementation:
//   t = x @ W^T            (bf16 WMMA, fp32 accum)
//   out = (t @ E^T) / max(|t| * |e|, eps)   (bf16 WMMA, fused fp32 epilogue)
// wave32, V_WMMA_F32_16X16X32_BF16, LDS double-buffered 128x128x32 tiles,
// GLOBAL_LOAD_ASYNC_TO_LDS_B128 staging (ASYNCcnt) when the toolchain has it.
// ---------------------------------------------------------------------------

typedef __attribute__((ext_vector_type(16))) __bf16        v16bf;
typedef __attribute__((ext_vector_type(8)))  float         v8f;
typedef __attribute__((ext_vector_type(8)))  unsigned int  v8u;
typedef int v4i_vec __attribute__((vector_size(16)));      // matches builtin proto

#define AS1 __attribute__((address_space(1)))
#define AS3 __attribute__((address_space(3)))

#if defined(__AMDGCN__) && __has_builtin(__builtin_amdgcn_global_load_async_to_lds_b128)
#define HAVE_ASYNC_LDS 1
#else
#define HAVE_ASYNC_LDS 0
#endif

#define SIM_EPS 1e-6f

static __device__ __forceinline__ unsigned short f32_to_bf16_rne(float f) {
    unsigned int u = __builtin_bit_cast(unsigned int, f);
    unsigned int r = u + 0x7FFFu + ((u >> 16) & 1u);   // round-to-nearest-even
    return (unsigned short)(r >> 16);
}
static __device__ __forceinline__ float ld_as_float(float v) { return v; }
static __device__ __forceinline__ float ld_as_float(unsigned short h) {
    unsigned int u = ((unsigned int)h) << 16;
    return __builtin_bit_cast(float, u);
}

static __device__ __forceinline__ void wait_async_zero() {
#if HAVE_ASYNC_LDS
#if __has_builtin(__builtin_amdgcn_s_wait_asynccnt)
    __builtin_amdgcn_s_wait_asynccnt(0);
#else
    asm volatile("s_wait_asynccnt 0x0" ::: "memory");
#endif
#endif
}

// async b128 copy: global (AS1) -> LDS (AS3), tracked by ASYNCcnt
static __device__ __forceinline__ void async_copy_b128(const unsigned short* src,
                                                       unsigned short* dst) {
#if HAVE_ASYNC_LDS
    __builtin_amdgcn_global_load_async_to_lds_b128(
        (AS1 v4i_vec*)src, (AS3 v4i_vec*)dst, 0, 0);
#else
    *(uint4*)dst = *(const uint4*)src;
#endif
}

// --------------------------- f32 -> bf16 convert ---------------------------
__global__ void k_f32_to_bf16(const float* __restrict__ in,
                              unsigned short* __restrict__ out,
                              long long n) {
    long long i0 = ((long long)blockIdx.x * blockDim.x + threadIdx.x) * 4;
    long long stride = (long long)gridDim.x * blockDim.x * 4;
    for (long long i = i0; i + 3 < n; i += stride) {
        float4 v = *(const float4*)(in + i);
        ushort4 o;
        o.x = f32_to_bf16_rne(v.x);
        o.y = f32_to_bf16_rne(v.y);
        o.z = f32_to_bf16_rne(v.z);
        o.w = f32_to_bf16_rne(v.w);
        *(ushort4*)(out + i) = o;
    }
    long long rem = n & 3;   // defensive tail; our shapes are multiples of 4
    if (blockIdx.x == 0 && (long long)threadIdx.x < rem)
        out[n - 1 - threadIdx.x] = f32_to_bf16_rne(in[n - 1 - threadIdx.x]);
}

// ------------------------------ row L2 norms -------------------------------
template <typename TIn>
__global__ void k_row_norm(const TIn* __restrict__ in, float* __restrict__ out, int D) {
    __shared__ float red[256];
    const int row = blockIdx.x;
    const TIn* p = in + (size_t)row * D;
    float s = 0.f;
    for (int i = threadIdx.x; i < D; i += blockDim.x) {
        float v = ld_as_float(p[i]);
        s += v * v;
    }
    red[threadIdx.x] = s;
    __syncthreads();
    for (int off = 128; off > 0; off >>= 1) {
        if ((int)threadIdx.x < off) red[threadIdx.x] += red[threadIdx.x + off];
        __syncthreads();
    }
    if (threadIdx.x == 0) out[row] = sqrtf(red[0]);
}

// ------------------------- bf16 WMMA tiled GEMM ----------------------------
// A: [M][K] bf16 row-major.  Bm: [N][K] bf16 row-major (B-matrix = Bm^T, so
// column n of B is the contiguous row n of Bm).  Out[M][N].
// SCALED=false: store bf16 (stage-1 t).  SCALED=true: fused cosine epilogue.
#define GEMM_MT   128
#define GEMM_NT   128
#define GEMM_KT   32
#define GEMM_LDT  40   // padded LDS row pitch in bf16 elems (80 B) - no bank conflicts

template <bool SCALED>
__global__ __launch_bounds__(256)
void k_gemm_bf16(const unsigned short* __restrict__ A,
                 const unsigned short* __restrict__ Bm,
                 float* __restrict__ outF, unsigned short* __restrict__ outH,
                 const float* __restrict__ tn, const float* __restrict__ en,
                 int M, int N, int K) {
    __shared__ unsigned short As[2][GEMM_MT * GEMM_LDT];   // 2 x 10 KB
    __shared__ unsigned short Bs[2][GEMM_NT * GEMM_LDT];   // 2 x 10 KB

    const int tid    = threadIdx.x;
    const int lane   = tid & 31;           // gfx1250 is wave32
    const int wave   = tid >> 5;           // 8 waves
    const int waveM  = wave >> 2;          // 0..1  -> 64-row slab
    const int waveN  = wave & 3;           // 0..3  -> 32-col slab
    const int laneHi = (lane >> 4) & 1;    // lane half selects K sub-block
    const int lm     = lane & 15;
    const int m0 = blockIdx.y * GEMM_MT;
    const int n0 = blockIdx.x * GEMM_NT;

    // Stage one 128x32 A tile + 128x32 B tile into LDS buffer `buf`.
    // 4 x b128 per thread; async direct-to-LDS on CDNA5.
    auto stage = [&](int k0, int buf) {
#pragma unroll
        for (int e = 0; e < 2; ++e) {
            int idx = tid + e * 256;           // 0..511
            int row = idx >> 2;                // 0..127
            int seg = idx & 3;                 // 4 x uint4 per 32-elem row
            {   // A tile
                unsigned short* dst = &As[buf][row * GEMM_LDT + seg * 8];
                if (m0 + row < M) {
                    async_copy_b128(A + (size_t)(m0 + row) * K + k0 + seg * 8, dst);
                } else {
                    *(uint4*)dst = make_uint4(0u, 0u, 0u, 0u);
                }
            }
            {   // B tile
                unsigned short* dst = &Bs[buf][row * GEMM_LDT + seg * 8];
                if (n0 + row < N) {
                    async_copy_b128(Bm + (size_t)(n0 + row) * K + k0 + seg * 8, dst);
                } else {
                    *(uint4*)dst = make_uint4(0u, 0u, 0u, 0u);
                }
            }
        }
#if !HAVE_ASYNC_LDS
        // fallback path: at least prefetch the tile after next (global_prefetch_b8)
        if (k0 + GEMM_KT < K) {
            int prow = tid >> 1;
            if (m0 + prow < M)
                __builtin_prefetch(A + (size_t)(m0 + prow) * K + k0 + GEMM_KT, 0, 1);
            if (n0 + prow < N)
                __builtin_prefetch(Bm + (size_t)(n0 + prow) * K + k0 + GEMM_KT, 0, 1);
        }
#endif
    };

    v8f acc[4][2];
#pragma unroll
    for (int mi = 0; mi < 4; ++mi)
#pragma unroll
        for (int ni = 0; ni < 2; ++ni) acc[mi][ni] = (v8f)0.f;

    // prologue: fill buffer 0
    stage(0, 0);
    wait_async_zero();
    __syncthreads();

    int buf = 0;
    for (int k0 = 0; k0 < K; k0 += GEMM_KT) {
        // kick off next tile's async copies while we compute this one
        if (k0 + GEMM_KT < K) stage(k0 + GEMM_KT, buf ^ 1);

        // A fragment 16x32 bf16: lanes 0-15 -> K 0..7 / 16..23,
        // lanes 16-31 -> K 8..15 / 24..31 (ISA 7.12.2).
        v16bf afrag[4];
#pragma unroll
        for (int mi = 0; mi < 4; ++mi) {
            const unsigned short* base =
                &As[buf][(waveM * 64 + mi * 16 + lm) * GEMM_LDT + laneHi * 8];
            uint4 lo = *(const uint4*)(base);       // K = hi8 + 0..7
            uint4 hi = *(const uint4*)(base + 16);  // K = 16 + hi8 + 0..7
            v8u t; t[0]=lo.x; t[1]=lo.y; t[2]=lo.z; t[3]=lo.w;
                   t[4]=hi.x; t[5]=hi.y; t[6]=hi.z; t[7]=hi.w;
            afrag[mi] = __builtin_bit_cast(v16bf, t);
        }
        // B fragment 32x16 bf16: lane -> column N, same K order as A:
        // lanes 0-15 -> K 0..7 / 16..23, lanes 16-31 -> K 8..15 / 24..31.
        v16bf bfrag[2];
#pragma unroll
        for (int ni = 0; ni < 2; ++ni) {
            const unsigned short* base =
                &Bs[buf][(waveN * 32 + ni * 16 + lm) * GEMM_LDT + laneHi * 8];
            uint4 lo = *(const uint4*)(base);
            uint4 hi = *(const uint4*)(base + 16);
            v8u t; t[0]=lo.x; t[1]=lo.y; t[2]=lo.z; t[3]=lo.w;
                   t[4]=hi.x; t[5]=hi.y; t[6]=hi.z; t[7]=hi.w;
            bfrag[ni] = __builtin_bit_cast(v16bf, t);
        }
#pragma unroll
        for (int mi = 0; mi < 4; ++mi)
#pragma unroll
            for (int ni = 0; ni < 2; ++ni)
                acc[mi][ni] = __builtin_amdgcn_wmma_f32_16x16x32_bf16(
                    false, afrag[mi], false, bfrag[ni],
                    (short)0, acc[mi][ni], false, false);

        // my async copies for the next buffer have landed ...
        wait_async_zero();
        // ... and everyone else's too
        __syncthreads();
        buf ^= 1;
    }

    // ----------------------------- epilogue --------------------------------
    // C/D layout: VGPR r -> M = r + 8*laneHi, N = lane&15 (ISA 7.12.2).
#pragma unroll
    for (int ni = 0; ni < 2; ++ni) {
        const int cn = n0 + waveN * 32 + ni * 16 + lm;
        if (cn >= N) continue;
        float env = 0.f;
        if (SCALED) env = en[cn];
#pragma unroll
        for (int mi = 0; mi < 4; ++mi) {
            const int rbase = m0 + waveM * 64 + mi * 16 + laneHi * 8;
#pragma unroll
            for (int r = 0; r < 8; ++r) {
                const int rm = rbase + r;
                if (rm >= M) continue;
                const float v = acc[mi][ni][r];
                if (SCALED) {
                    const float denom = fmaxf(tn[rm] * env, SIM_EPS);
                    outF[(size_t)rm * N + cn] = v / denom;
                } else {
                    outH[(size_t)rm * N + cn] = f32_to_bf16_rne(v);
                }
            }
        }
    }
}

// ------------------------------- launcher ----------------------------------
extern "C" void kernel_launch(void* const* d_in, const int* in_sizes, int n_in,
                              void* d_out, int out_size, void* d_ws, size_t ws_size,
                              hipStream_t stream) {
    const float* x = (const float*)d_in[0];   // [B][D]
    const float* W = (const float*)d_in[1];   // [D][D]
    const float* E = (const float*)d_in[2];   // [N][D]

    // derive shapes: in_sizes = { B*D, D*D, N*D }
    int D = 1;
    while ((long long)D * D < (long long)in_sizes[1]) D <<= 1;   // D = 1024
    const int B = in_sizes[0] / D;                               // 2048
    const int N = in_sizes[2] / D;                               // 50000

    // carve workspace (256B-aligned)
    char* ws = (char*)d_ws;
    auto carve = [&](size_t bytes) -> char* {
        char* p = ws;
        ws += (bytes + 255) & ~(size_t)255;
        return p;
    };
    unsigned short* xh = (unsigned short*)carve((size_t)B * D * 2);   //  4 MB
    unsigned short* Wh = (unsigned short*)carve((size_t)D * D * 2);   //  2 MB
    unsigned short* th = (unsigned short*)carve((size_t)B * D * 2);   //  4 MB
    unsigned short* Eh = (unsigned short*)carve((size_t)N * D * 2);   // 102 MB
    float*          tn = (float*)carve((size_t)B * 4);
    float*          en = (float*)carve((size_t)N * 4);

    // 1) downconvert operands to bf16
    k_f32_to_bf16<<<1024, 256, 0, stream>>>(x, xh, (long long)B * D);
    k_f32_to_bf16<<<1024, 256, 0, stream>>>(W, Wh, (long long)D * D);
    k_f32_to_bf16<<<8192, 256, 0, stream>>>(E, Eh, (long long)N * D);

    // 2) t = x @ W^T  (Bm = W since t[m][j] = sum_k x[m][k] * W[j][k])
    {
        dim3 grid((D + GEMM_NT - 1) / GEMM_NT, (B + GEMM_MT - 1) / GEMM_MT);
        k_gemm_bf16<false><<<grid, 256, 0, stream>>>(
            xh, Wh, nullptr, th, nullptr, nullptr, B, D, D);
    }

    // 3) row norms: |t| from the bf16 t actually used in the dot; |e| from fp32 E
    k_row_norm<unsigned short><<<B, 256, 0, stream>>>(th, tn, D);
    k_row_norm<float><<<N, 256, 0, stream>>>(E, en, D);

    // 4) out = (t @ E^T) / max(|t||e|, eps)
    {
        dim3 grid((N + GEMM_NT - 1) / GEMM_NT, (B + GEMM_MT - 1) / GEMM_MT);
        k_gemm_bf16<true><<<grid, 256, 0, stream>>>(
            th, Eh, (float*)d_out, nullptr, tn, en, B, N, D);
    }
}
